// TemporalAttentionEncoder_90091234001358
// MI455X (gfx1250) — matrix operand order repaired and checked by reading, not compiled, not run on hardware
//
#include <hip/hip_runtime.h>
#include <hip/hip_bf16.h>
#include <math.h>
#include <stdint.h>

typedef __attribute__((ext_vector_type(16))) _Float16 v16h;
typedef __attribute__((ext_vector_type(8)))  _Float16 v8h;
typedef __attribute__((ext_vector_type(4)))  _Float16 v4h;
typedef __attribute__((ext_vector_type(8)))  float    v8f;
typedef __attribute__((ext_vector_type(4)))  float    v4f;

#define B_TOT 4096
#define SEQ   64
#define DIM   128
#define HQ    4
#define DKQ   32

static __device__ __forceinline__ v16h cat16(v8h lo, v8h hi) {
  return __builtin_shufflevector(lo, hi, 0, 1, 2, 3, 4, 5, 6, 7,
                                         8, 9, 10, 11, 12, 13, 14, 15);
}

// Async global->LDS copy (16B per lane), CDNA5 GLOBAL_LOAD_ASYNC_TO_LDS_B128,
// GVS mode: mem = SGPR64 base + VGPR32 offset; VDST = LDS byte offset.
static __device__ __forceinline__
void async_copy_b128(uint32_t lds_off, const void* gbase, uint32_t goff) {
  asm volatile("global_load_async_to_lds_b128 %0, %1, %2"
               :
               : "v"(lds_off), "v"(goff),
                 "s"((unsigned long long)(uintptr_t)gbase)
               : "memory");
}
static __device__ __forceinline__ void wait_async0() {
  asm volatile("s_wait_asynccnt 0" ::: "memory");
}
static __device__ __forceinline__ uint32_t lds_offset_of(const void* p) {
  return (uint32_t)(uintptr_t)p;   // generic LDS ptr: low 32 bits = LDS offset
}

// ---------------------------------------------------------------------------
// prep: PE table, folded q-path weights (Wqc = w_q@w2), BN-folded MLP weights,
//       PRE-TRANSPOSED f16 copies of all WMMA B-matrices ([n][k] layout).
// ---------------------------------------------------------------------------
__global__ __launch_bounds__(256)
void prep_kernel(const float* __restrict__ w_q, const float* __restrict__ b_q,
                 const float* __restrict__ w_k,
                 const float* __restrict__ w2,  const float* __restrict__ b2,
                 const float* __restrict__ w3a, const float* __restrict__ b3a,
                 const float* __restrict__ g3a, const float* __restrict__ be3a,
                 const float* __restrict__ w3b, const float* __restrict__ b3b,
                 const float* __restrict__ g3b, const float* __restrict__ be3b,
                 float* __restrict__ pe, float* __restrict__ Wqc,
                 float* __restrict__ bqc, float* __restrict__ b3a_s,
                 float* __restrict__ b3b_s, _Float16* __restrict__ wkT_h,
                 _Float16* __restrict__ w3aT_h, _Float16* __restrict__ w3bT_h)
{
  const int gtid = blockIdx.x * blockDim.x + threadIdx.x;
  const int gs   = gridDim.x * blockDim.x;

  // interleaved sin/cos positional encoding, T=1000
  for (int i = gtid; i < SEQ * DIM; i += gs) {
    int s = i >> 7, d = i & 127;
    int pair = d >> 1;
    float div = powf(1000.0f, (float)(2 * pair) * (1.0f / 128.0f));
    float ang = (float)s / div;
    pe[i] = (d & 1) ? cosf(ang) : sinf(ang);
  }
  // Wqc = w_q @ w2  (mean commutes with linear layers -> q path collapses)
  for (int i = gtid; i < DIM * DIM; i += gs) {
    int r = i >> 7, c = i & 127;
    float a = 0.f;
    for (int m = 0; m < DIM; ++m) a = fmaf(w_q[r * DIM + m], w2[m * DIM + c], a);
    Wqc[i] = a;
  }
  for (int j = gtid; j < DIM; j += gs) {
    float a = b2[j];
    for (int m = 0; m < DIM; ++m) a = fmaf(b_q[m], w2[m * DIM + j], a);
    bqc[j]   = a;
    b3a_s[j] = b3a[j] * g3a[j] + be3a[j];   // BN folded into bias
    b3b_s[j] = b3b[j] * g3b[j] + be3b[j];
  }
  // transposed f16 weights: [n][k] so kernel stage-in is a contiguous copy
  for (int i = gtid; i < DIM * DIM; i += gs) {
    int n = i >> 7, k = i & 127;
    wkT_h[i] = (_Float16)w_k[k * DIM + n];
  }
  for (int i = gtid; i < HQ * DIM * DIM; i += gs) {
    int n = i >> 9, k = i & 511;            // [128][512]
    w3aT_h[i] = (_Float16)(w3a[k * DIM + n] * g3a[n]);
  }
  for (int i = gtid; i < DIM * DIM; i += gs) {
    int n = i >> 7, k = i & 127;
    w3bT_h[i] = (_Float16)(w3b[k * DIM + n] * g3b[n]);
  }
}

// ---------------------------------------------------------------------------
// main attention kernel: one workgroup (8 wave32) per batch element.
//  - w_k^T tile staged via GLOBAL_LOAD_ASYNC_TO_LDS_B128 (ASYNCcnt)
//  - e_p tile -> LDS f16 via float4 loads (+fused f32 column means)
//  - K = e_p @ w_k + b_k via v_wmma_f32_16x16x32_f16 (2x ds_load_b128/frag)
//  - f32 logits + softmax; attn over permuted-batch v (faithful to reference)
// ---------------------------------------------------------------------------
__global__ __launch_bounds__(256)
void attn_kernel(const float* __restrict__ x, const float* __restrict__ b_k,
                 const float* __restrict__ pe, const float* __restrict__ Wqc,
                 const float* __restrict__ bqc, const _Float16* __restrict__ wkT_h,
                 _Float16* __restrict__ attn_h)
{
  __shared__ _Float16 sA[SEQ * 136];   // e_p[b] f16, row-major, padded
  __shared__ _Float16 sBT[DIM * 136];  // w_k^T [n][k], padded
  __shared__ float    sK[SEQ * 132];   // K result f32
  __shared__ float    sBk[DIM];
  __shared__ float    sEm[DIM];        // mean_s e_p
  __shared__ float    sQh[DIM];        // q_hat row
  __shared__ float    sS[HQ * SEQ];    // probs
  __shared__ float    sRed[HQ * SEQ];  // reduction scratch
  __shared__ float    sP8[8 * DIM];    // 8-way column partial sums

  const int b    = blockIdx.x;
  const int tid  = threadIdx.x;
  const int srow = tid >> 5;           // wave id -> row phase (s = srow + 8*i)
  const int c4   = (tid & 31) * 4;     // 16-byte column chunk

  // async stage w_k^T: 2048 16B chunks; dst honors the padded LDS stride
  {
    const uint32_t base = lds_offset_of(&sBT[0]);
#pragma unroll
    for (int j = 0; j < 8; ++j) {
      int i  = tid + 256 * j;          // chunk index
      int n  = i >> 4, ch = i & 15;    // row n, 16B chunk within row
      async_copy_b128(base + (uint32_t)(n * 136 + ch * 8) * 2,
                      wkT_h, (uint32_t)i * 16);
    }
  }
  if (tid < DIM) sBk[tid] = b_k[tid];

  // e_p tile (float4 coalesced) + fused column partial sums; cache pe in regs
  v4f pe4[8];
  {
    const float* xb = x + (size_t)b * SEQ * DIM;
    v4f acc = {};
#pragma unroll
    for (int i = 0; i < 8; ++i) {
      const int s = srow + 8 * i;
      v4f xv = *(const v4f*)(xb + s * DIM + c4);
      pe4[i] = *(const v4f*)(pe + s * DIM + c4);
      v4f v  = xv + pe4[i];
      *(v4h*)&sA[s * 136 + c4] = __builtin_convertvector(v, v4h);
      acc += v;
    }
#pragma unroll
    for (int j = 0; j < 4; ++j) sP8[srow * DIM + c4 + j] = acc[j];
  }
  // prefetch the 4 permuted-batch x tiles for the attn pass (1 line/thread)
#pragma unroll
  for (int h2 = 0; h2 < HQ; ++h2) {
    const int bs = (4 * b + h2) & (B_TOT - 1);
    __builtin_prefetch(x + (size_t)bs * SEQ * DIM + tid * 32, 0, 0);
  }
  wait_async0();                       // sBT resident before barrier
  __syncthreads();
  if (tid < DIM) {
    float a = 0.f;
#pragma unroll
    for (int r = 0; r < 8; ++r) a += sP8[r * DIM + tid];
    sEm[tid] = a * (1.0f / 64.0f);
  }
  __syncthreads();

  // q_hat = em @ Wqc + bqc (tiny matvec, f32)
  if (tid < DIM) {
    float a = bqc[tid];
    for (int i = 0; i < DIM; ++i) a = fmaf(sEm[i], Wqc[i * DIM + tid], a);
    sQh[tid] = a;
  }

  // ---- WMMA: K = e_p @ w_k + b_k  (M=64, N=128, K=128) ----
  {
    const int w    = tid >> 5;
    const int lane = tid & 31;
    const int lrow = lane & 15;
    const int lkh  = lane >> 4;
    const int mt   = w >> 1;          // 4 M-tiles
    const int ntb  = (w & 1) * 4;     // 8 N-tiles split 4/4

    v16h afrag[4];
#pragma unroll
    for (int kt = 0; kt < 4; ++kt) {
      // ISA 16-bit A 16x32: lane kh half: halves [8kh..8kh+7] and [16+8kh..]
      const _Float16* ap = sA + (mt * 16 + lrow) * 136 + kt * 32 + 8 * lkh;
      v8h a0 = *(const v8h*)ap;
      v8h a1 = *(const v8h*)(ap + 16);
      afrag[kt] = cat16(a0, a1);
    }
#pragma unroll
    for (int nt = 0; nt < 4; ++nt) {
      const int n0 = (ntb + nt) * 16;
      v8f c = {};
#pragma unroll
      for (int kt = 0; kt < 4; ++kt) {
        // B 32x16: lane = N, K = [8kh..8kh+7] and [16+8kh..] in sBT[n][k]
        const _Float16* bp = sBT + (n0 + lrow) * 136 + kt * 32 + lkh * 8;
        v8h b0 = *(const v8h*)bp;
        v8h b1 = *(const v8h*)(bp + 16);
        c = __builtin_amdgcn_wmma_f32_16x16x32_f16(
                false, afrag[kt], false, cat16(b0, b1), (short)0, c, false, false);
      }
#pragma unroll
      for (int r = 0; r < 8; ++r) {   // C/D: row = r + 8*(lane>>4)
        int row = mt * 16 + r + 8 * lkh;
        int col = n0 + lrow;
        sK[row * 132 + col] = c[r] + sBk[col];
      }
    }
  }
  __syncthreads();

  // ---- scores + softmax (f32); tid -> (h = tid>>6, s = tid&63) ----
  const int h  = tid >> 6;
  const int sI = tid & 63;
  float sc = 0.f;
  {
    const int base = h * DKQ;
#pragma unroll
    for (int q4 = 0; q4 < DKQ / 4; ++q4) {
      v4f qv = *(const v4f*)&sQh[base + q4 * 4];
      v4f kv = *(const v4f*)&sK[sI * 132 + base + q4 * 4];
      sc = fmaf(qv[0], kv[0], sc);
      sc = fmaf(qv[1], kv[1], sc);
      sc = fmaf(qv[2], kv[2], sc);
      sc = fmaf(qv[3], kv[3], sc);
    }
    sc *= 0.1767766952966369f;        // 1/sqrt(32)
  }
  sRed[tid] = sc;
  __syncthreads();
  for (int st = 32; st > 0; st >>= 1) {
    if (sI < st) sRed[h * SEQ + sI] = fmaxf(sRed[h * SEQ + sI], sRed[h * SEQ + sI + st]);
    __syncthreads();
  }
  const float mx = sRed[h * SEQ];
  __syncthreads();
  const float ex = __expf(sc - mx);
  sRed[tid] = ex;
  __syncthreads();
  for (int st = 32; st > 0; st >>= 1) {
    if (sI < st) sRed[h * SEQ + sI] += sRed[h * SEQ + sI + st];
    __syncthreads();
  }
  sS[tid] = ex / sRed[h * SEQ];
  __syncthreads();

  // ---- attn: v = tile(e_p) head-major -> batch (4b+h) % B (faithful) ----
  for (int h2 = 0; h2 < HQ; ++h2) {
    const int bs = (4 * b + h2) & (B_TOT - 1);
    const float* xv = x + (size_t)bs * SEQ * DIM;   // L2 hit: x fits in 192MB L2
    v4f a4 = {};
#pragma unroll
    for (int i = 0; i < 8; ++i) {
      const int s = srow + 8 * i;                   // wave-uniform row
      const float p = sS[h2 * SEQ + s];
      v4f pv = {p, p, p, p};
      v4f xx = *(const v4f*)(xv + s * DIM + c4);
      a4 += pv * (xx + pe4[i]);
    }
#pragma unroll
    for (int j = 0; j < 4; ++j) sP8[srow * DIM + c4 + j] = a4[j];
    __syncthreads();
    if (tid < DIM) {
      float a = 0.f;
#pragma unroll
      for (int r = 0; r < 8; ++r) a += sP8[r * DIM + tid];
      attn_h[(size_t)b * (HQ * DIM) + h2 * DIM + tid] = (_Float16)a;
    }
    __syncthreads();
  }
}

// ---------------------------------------------------------------------------
// MLP: (4096x512)@(512x128) -> ReLU -> (4096x128)@(128x128) -> ReLU
// 64 rows per workgroup; attn tile async-staged into LDS; pre-transposed
// weight tiles staged with contiguous v8h copies -> 2x ds_load_b128/fragment.
// ---------------------------------------------------------------------------
__global__ __launch_bounds__(256)
void mlp_kernel(const _Float16* __restrict__ attn_h, const _Float16* __restrict__ w3aT_h,
                const float* __restrict__ b3a_s, const _Float16* __restrict__ w3bT_h,
                const float* __restrict__ b3b_s, float* __restrict__ out)
{
  __shared__ _Float16 sAt[64 * 520];   // attn rows, f16, padded
  __shared__ _Float16 sBT[DIM * 40];   // transposed weight k-tile [n][k<32]
  __shared__ _Float16 sH[64 * 136];    // hidden layer, f16

  const int tid     = threadIdx.x;
  const int rowBase = blockIdx.x * 64;
  const int w    = tid >> 5;
  const int lane = tid & 31;
  const int lrow = lane & 15;
  const int lkh  = lane >> 4;
  const int mt   = w >> 1;
  const int ntb  = (w & 1) * 4;

  // async stage attn tile: 64 rows x 512 halves = 4096 16B chunks
  {
    const uint32_t base = lds_offset_of(&sAt[0]);
    const uint32_t gbase_off = (uint32_t)rowBase * 512u * 2u;
#pragma unroll
    for (int j = 0; j < 16; ++j) {
      int i = tid + 256 * j;
      int row = i >> 6, ch = i & 63;
      async_copy_b128(base + (uint32_t)(row * 520 + ch * 8) * 2,
                      attn_h, gbase_off + (uint32_t)i * 16);
    }
  }

  v8f zero = {};
  v8f acc[4] = {zero, zero, zero, zero};
  bool first = true;

  // layer 1: K = 512 -> 16 k-tiles
  for (int kt = 0; kt < 16; ++kt) {
    __syncthreads();
    // stage w3a^T k-tile: 128 rows x 32 halves = 512 16B chunks
    for (int i = tid; i < 512; i += 256) {
      int n = i >> 2, ch = i & 3;
      *(v8h*)&sBT[n * 40 + ch * 8] =
          *(const v8h*)&w3aT_h[(size_t)n * 512 + kt * 32 + ch * 8];
    }
    if (first) { wait_async0(); first = false; }   // sAt resident
    __syncthreads();
    const _Float16* ap = sAt + (mt * 16 + lrow) * 520 + kt * 32 + 8 * lkh;
    v8h a0 = *(const v8h*)ap;
    v8h a1 = *(const v8h*)(ap + 16);
    v16h af = cat16(a0, a1);
#pragma unroll
    for (int nt = 0; nt < 4; ++nt) {
      const _Float16* bp = sBT + ((ntb + nt) * 16 + lrow) * 40 + lkh * 8;
      v8h b0 = *(const v8h*)bp;
      v8h b1 = *(const v8h*)(bp + 16);
      acc[nt] = __builtin_amdgcn_wmma_f32_16x16x32_f16(
                    false, af, false, cat16(b0, b1), (short)0, acc[nt], false, false);
    }
  }
  __syncthreads();
  // bias + ReLU -> sH (f16)
#pragma unroll
  for (int nt = 0; nt < 4; ++nt) {
    int col = (ntb + nt) * 16 + lrow;
    float bb = b3a_s[col];
#pragma unroll
    for (int r = 0; r < 8; ++r) {
      int row = mt * 16 + r + 8 * lkh;
      float v = acc[nt][r] + bb;
      sH[row * 136 + col] = (_Float16)(v > 0.f ? v : 0.f);
    }
  }

  // layer 2: K = 128 -> 4 k-tiles
  v8f acc2[4] = {zero, zero, zero, zero};
  for (int kt = 0; kt < 4; ++kt) {
    __syncthreads();
    for (int i = tid; i < 512; i += 256) {   // stage w3b^T k-tile
      int n = i >> 2, ch = i & 3;
      *(v8h*)&sBT[n * 40 + ch * 8] =
          *(const v8h*)&w3bT_h[(size_t)n * 128 + kt * 32 + ch * 8];
    }
    __syncthreads();
    const _Float16* ap = sH + (mt * 16 + lrow) * 136 + kt * 32 + 8 * lkh;
    v8h a0 = *(const v8h*)ap;
    v8h a1 = *(const v8h*)(ap + 16);
    v16h af = cat16(a0, a1);
#pragma unroll
    for (int nt = 0; nt < 4; ++nt) {
      const _Float16* bp = sBT + ((ntb + nt) * 16 + lrow) * 40 + lkh * 8;
      v8h b0 = *(const v8h*)bp;
      v8h b1 = *(const v8h*)(bp + 16);
      acc2[nt] = __builtin_amdgcn_wmma_f32_16x16x32_f16(
                     false, af, false, cat16(b0, b1), (short)0, acc2[nt], false, false);
    }
  }
#pragma unroll
  for (int nt = 0; nt < 4; ++nt) {
    int col = (ntb + nt) * 16 + lrow;
    float bb = b3b_s[col];
#pragma unroll
    for (int r = 0; r < 8; ++r) {
      int row = mt * 16 + r + 8 * lkh;
      float v = acc2[nt][r] + bb;
      out[(size_t)(rowBase + row) * DIM + col] = v > 0.f ? v : 0.f;
    }
  }
}

// ---------------------------------------------------------------------------
extern "C" void kernel_launch(void* const* d_in, const int* in_sizes, int n_in,
                              void* d_out, int out_size, void* d_ws, size_t ws_size,
                              hipStream_t stream) {
  const float* x    = (const float*)d_in[0];
  const float* w_q  = (const float*)d_in[1];
  const float* b_q  = (const float*)d_in[2];
  const float* w_k  = (const float*)d_in[3];
  const float* b_k  = (const float*)d_in[4];
  const float* w2   = (const float*)d_in[5];
  const float* b2   = (const float*)d_in[6];
  const float* w3a  = (const float*)d_in[7];
  const float* b3a  = (const float*)d_in[8];
  const float* g3a  = (const float*)d_in[9];
  const float* be3a = (const float*)d_in[10];
  const float* w3b  = (const float*)d_in[11];
  const float* b3b  = (const float*)d_in[12];
  const float* g3b  = (const float*)d_in[13];
  const float* be3b = (const float*)d_in[14];
  float* out = (float*)d_out;

  // workspace carve-up (~4.5 MiB total)
  char* p = (char*)d_ws;
  auto carve = [&](size_t bytes) {
    void* r = (void*)p;
    p += (bytes + 255) & ~(size_t)255;
    return r;
  };
  float*    pe     = (float*)carve(SEQ * DIM * sizeof(float));
  float*    Wqc    = (float*)carve(DIM * DIM * sizeof(float));
  float*    bqc    = (float*)carve(DIM * sizeof(float));
  float*    b3a_s  = (float*)carve(DIM * sizeof(float));
  float*    b3b_s  = (float*)carve(DIM * sizeof(float));
  _Float16* wkT_h  = (_Float16*)carve(DIM * DIM * sizeof(_Float16));
  _Float16* w3aT_h = (_Float16*)carve((size_t)HQ * DIM * DIM * sizeof(_Float16));
  _Float16* w3bT_h = (_Float16*)carve(DIM * DIM * sizeof(_Float16));
  _Float16* attn_h = (_Float16*)carve((size_t)B_TOT * HQ * DIM * sizeof(_Float16));

  prep_kernel<<<64, 256, 0, stream>>>(w_q, b_q, w_k, w2, b2, w3a, b3a, g3a, be3a,
                                      w3b, b3b, g3b, be3b,
                                      pe, Wqc, bqc, b3a_s, b3b_s, wkT_h, w3aT_h, w3bT_h);
  attn_kernel<<<B_TOT, 256, 0, stream>>>(x, b_k, pe, Wqc, bqc, wkT_h, attn_h);
  mlp_kernel<<<B_TOT / 64, 256, 0, stream>>>(attn_h, w3aT_h, b3a_s, w3bT_h, b3b_s, out);
}
